// RPRefineDetMultiBoxLoss_23356032156117
// MI455X (gfx1250) — compile-verified, matrix operand order checked by reading, not run on hardware
//
#include <hip/hip_runtime.h>
#include <hip/hip_bf16.h>
#include <hip/hip_fp16.h>

#define NCLS 81
#define B_   64
#define P_   16320
#define O_   24
#define THRESH  0.5f
#define THETA   0.01f
#define NEGPOS  3

typedef __attribute__((ext_vector_type(16))) _Float16 v16h;
typedef __attribute__((ext_vector_type(8)))  float    v8f;

__device__ __forceinline__ float4 ldg4(const float* p) { return *(const float4*)p; }

__device__ __forceinline__ float blockReduceSum(float v, float* sbuf) {
    int tid = threadIdx.x;
    sbuf[tid] = v; __syncthreads();
    for (int s = blockDim.x >> 1; s > 0; s >>= 1) {
        if (tid < s) sbuf[tid] += sbuf[tid + s];
        __syncthreads();
    }
    float r = sbuf[0]; __syncthreads();
    return r;
}

// ---------------- K0: zero accumulator region of workspace ----------------
__global__ void init_kernel(unsigned* words, int n) {
    for (int i = blockIdx.x * blockDim.x + threadIdx.x; i < n; i += gridDim.x * blockDim.x)
        words[i] = 0u;
}

// ---------------- K1: decode ARM boxes, IoU match ----------------
__global__ void match_kernel(const float* __restrict__ arm_loc,
                             const float* __restrict__ priors,
                             const float* __restrict__ gt_boxes,
                             float* __restrict__ bto, int* __restrict__ bti,
                             unsigned long long* __restrict__ gBest)
{
    __shared__ float sT[O_ * 4];
    __shared__ unsigned long long sBest[O_];
    const int b = blockIdx.x;
    const int tid = threadIdx.x;

    // Async copy of GT boxes (96 dwords) into LDS: CDNA5 async global->LDS path.
    if (tid < O_ * 4) {
        unsigned long long ga =
            (unsigned long long)(size_t)(gt_boxes + (size_t)b * O_ * 4 + tid);
        unsigned ldsOff = (unsigned)(size_t)(&sT[tid]);
        asm volatile("global_load_async_to_lds_b32 %0, %1, off"
                     :: "v"(ldsOff), "v"(ga) : "memory");
    }
    if (tid < O_) sBest[tid] = 0ull;
    asm volatile("s_wait_asynccnt 0" ::: "memory");
    __syncthreads();

    for (int p = tid; p < P_; p += blockDim.x) {
        float4 pr = ldg4(priors + (size_t)p * 4);
        float4 l  = ldg4(arm_loc + ((size_t)b * P_ + p) * 4);
        float cx = pr.x + l.x * 0.1f * pr.z;
        float cy = pr.y + l.y * 0.1f * pr.w;
        float w  = pr.z * __expf(l.z * 0.2f);
        float h  = pr.w * __expf(l.w * 0.2f);
        float x0 = cx - 0.5f * w, y0 = cy - 0.5f * h;
        float x1 = cx + 0.5f * w, y1 = cy + 0.5f * h;
        float areaB = w * h;
        float best = -1.f; int bestO = 0;
        for (int o = 0; o < O_; ++o) {
            float a0 = sT[o*4], a1 = sT[o*4+1], a2 = sT[o*4+2], a3 = sT[o*4+3];
            float ix = fmaxf(fminf(a2, x1) - fmaxf(a0, x0), 0.f);
            float iy = fmaxf(fminf(a3, y1) - fmaxf(a1, y0), 0.f);
            float inter = ix * iy;
            float areaA = (a2 - a0) * (a3 - a1);
            float iou = inter / (areaA + areaB - inter);
            if (iou > best) { best = iou; bestO = o; }
            unsigned long long key =
                ((unsigned long long)__float_as_uint(iou) << 32) | (unsigned)p;
            atomicMax(&sBest[o], key);
        }
        bto[(size_t)b * P_ + p] = best;
        bti[(size_t)b * P_ + p] = bestO;
    }
    __syncthreads();
    if (tid < O_) atomicMax(&gBest[b * O_ + tid], sBest[tid]);
}

// ---------------- K2: per-GT best-prior override (ordered, last wins) ----------------
__global__ void override_kernel(const unsigned long long* __restrict__ gBest,
                                float* __restrict__ bto, int* __restrict__ bti)
{
    int b = blockIdx.x * blockDim.x + threadIdx.x;
    if (b >= B_) return;
    for (int o = 0; o < O_; ++o) {
        unsigned p = (unsigned)(gBest[b * O_ + o] & 0xffffffffu);
        bto[(size_t)b * P_ + p] = 2.0f;
        bti[(size_t)b * P_ + p] = o;
    }
}

// ---------------- K3: pos mask, conf_t, fused SmoothL1 loc loss ----------------
__global__ void pos_loc_kernel(const float* __restrict__ arm_loc,
                               const float* __restrict__ arm_conf,
                               const float* __restrict__ odm_loc,
                               const float* __restrict__ priors,
                               const float* __restrict__ gt_boxes,
                               const int*   __restrict__ gt_labels,
                               const float* __restrict__ bto,
                               const int*   __restrict__ bti,
                               int* __restrict__ conf_t, int* __restrict__ posA,
                               unsigned* __restrict__ num_pos,
                               float* __restrict__ scalars)
{
    __shared__ float sred[256];
    float lsum = 0.f;
    const long long total = (long long)B_ * P_;
    for (long long i = (long long)blockIdx.x * blockDim.x + threadIdx.x; i < total;
         i += (long long)gridDim.x * blockDim.x) {
        int b = (int)(i / P_), p = (int)(i % P_);
        float ov = bto[i]; int ti = bti[i];
        int ct = (ov < THRESH) ? 0 : gt_labels[b * O_ + ti];
        conf_t[i] = ct;
        float c0 = arm_conf[i * 2], c1 = arm_conf[i * 2 + 1];
        float s1 = 1.f / (1.f + __expf(c0 - c1));     // softmax class-1 score
        int ps = (ct > 0) && (s1 > THETA);
        posA[i] = ps;
        if (ps) {
            atomicAdd(&num_pos[b], 1u);
            float4 pr = ldg4(priors + (size_t)p * 4);
            float4 l  = ldg4(arm_loc + i * 4);
            float cx = pr.x + l.x * 0.1f * pr.z;
            float cy = pr.y + l.y * 0.1f * pr.w;
            float w  = pr.z * __expf(l.z * 0.2f);
            float h  = pr.w * __expf(l.w * 0.2f);
            const float* m = gt_boxes + ((size_t)b * O_ + ti) * 4;
            float mcx = 0.5f * (m[0] + m[2]), mcy = 0.5f * (m[1] + m[3]);
            float mw = m[2] - m[0], mh = m[3] - m[1];
            float g0 = (mcx - cx) / (0.1f * w);
            float g1 = (mcy - cy) / (0.1f * h);
            float g2 = __logf(mw / w) * 5.0f;   // /0.2
            float g3 = __logf(mh / h) * 5.0f;
            float4 od = ldg4(odm_loc + i * 4);
            float d[4] = { od.x - g0, od.y - g1, od.z - g2, od.w - g3 };
            #pragma unroll
            for (int k = 0; k < 4; ++k) {
                float ad = fabsf(d[k]);
                lsum += (ad < 1.f) ? 0.5f * ad * ad : ad - 0.5f;
            }
        }
    }
    float bs = blockReduceSum(lsum, sred);
    if (threadIdx.x == 0) atomicAdd(&scalars[0], bs);
}

// ---------------- K4: CE via WMMA class-sum reduction ----------------
// Wave handles 16 priors (one WMMA row group). Lane L owns row r=L&15,
// K-half h=L>>4 per the 16-bit A-matrix layout. exp(x-max) summed over
// classes by D = A x Ones + C with V_WMMA_F32_16X16X32_F16.
__global__ void ce_wmma_kernel(const float* __restrict__ odm_conf,
                               const int*   __restrict__ conf_t,
                               const int*   __restrict__ posA,
                               float* __restrict__ ce_mine,
                               float* __restrict__ scalars)
{
    __shared__ float sred[256];
    const int lane = threadIdx.x & 31;
    const int r = lane & 15;
    const int h = lane >> 4;
    const int flat = blockIdx.x * blockDim.x + threadIdx.x;
    const int gwave = flat >> 5;
    const int nWaves = (gridDim.x * blockDim.x) >> 5;
    const int nGroups = (B_ * P_) / 16;   // P_ % 16 == 0

    v16h ones;
    #pragma unroll
    for (int i = 0; i < 16; ++i) ones[i] = (_Float16)1.0f;

    float posCe = 0.f;

    for (int g = gwave; g < nGroups; g += nWaves) {
        const long long row = (long long)g * 16 + r;
        const float* crow = odm_conf + (size_t)row * NCLS;

        if (g + nWaves < nGroups)
            __builtin_prefetch(odm_conf + ((size_t)(g + nWaves) * 16 + r) * NCLS, 0, 1);

        // Row max: this lane covers classes with ((c>>3)&1)==h, partner has rest.
        float mx = -1e30f;
        for (int c0 = h * 8; c0 < NCLS; c0 += 16) {
            #pragma unroll
            for (int j = 0; j < 8; ++j) {
                int cc = c0 + j;
                if (cc < NCLS) mx = fmaxf(mx, crow[cc]);
            }
        }
        mx = fmaxf(mx, __shfl_xor(mx, 16, 32));

        int tgt = conf_t[row];
        float gathered = crow[tgt];
        int p = posA[row];

        v8f acc = {};
        #pragma unroll
        for (int ck = 0; ck < 3; ++ck) {
            v16h a;
            #pragma unroll
            for (int i = 0; i < 16; ++i) {
                int kk = (i < 8) ? (h * 8 + i) : (16 + h * 8 + (i - 8));
                int cc = ck * 32 + kk;
                float e = (cc < NCLS) ? __expf(crow[cc] - mx) : 0.f;
                a[i] = (_Float16)e;
            }
            acc = __builtin_amdgcn_wmma_f32_16x16x32_f16(
                false, a, false, ones, (short)0, acc, false, false);
        }

        // C/D layout: lanes 0-15 hold rows 0-7 (acc[j]=D[j][lane]),
        // lanes 16-31 hold rows 8-15. Writers chosen so own-row state matches.
        if (lane < 8) {
            float s = acc[lane];                    // row = lane = r
            float ce = (mx + __logf(s)) - gathered;
            ce_mine[row] = p ? 0.f : ce;
            if (p) posCe += ce;
        } else if (lane >= 24) {
            float s = acc[lane - 24];               // row = r = lane-16
            float ce = (mx + __logf(s)) - gathered;
            ce_mine[row] = p ? 0.f : ce;
            if (p) posCe += ce;
        }
    }
    float bs = blockReduceSum(posCe, sred);
    if (threadIdx.x == 0) atomicAdd(&scalars[1], bs);
}

// ---------------- K5: exact top-K sum per image (hard negative mining) ----------------
__global__ void topk_kernel(const float* __restrict__ ce_mine,
                            const unsigned* __restrict__ num_pos,
                            float* __restrict__ scalars)
{
    __shared__ int sCount;
    __shared__ float sSum;
    const int b = blockIdx.x;
    const float* v = ce_mine + (size_t)b * P_;
    int K = min(NEGPOS * (int)num_pos[b], P_ - 1);
    float result = 0.f;
    if (K > 0) {
        unsigned lo = 0u, hi = 0x7f800000u;   // values are >= 0
        while (lo < hi) {
            unsigned mid = lo + ((hi - lo) >> 1);
            float t = __uint_as_float(mid);
            if (threadIdx.x == 0) sCount = 0;
            __syncthreads();
            int c = 0;
            for (int i = threadIdx.x; i < P_; i += blockDim.x) c += (v[i] > t);
            atomicAdd(&sCount, c);
            __syncthreads();
            int total = sCount;
            __syncthreads();
            if (total >= K) lo = mid + 1; else hi = mid;
        }
        float tstar = __uint_as_float(lo);    // K-th largest value
        if (threadIdx.x == 0) { sCount = 0; sSum = 0.f; }
        __syncthreads();
        int c = 0; float s = 0.f;
        for (int i = threadIdx.x; i < P_; i += blockDim.x) {
            float x = v[i];
            if (x > tstar) { ++c; s += x; }
        }
        atomicAdd(&sCount, c); atomicAdd(&sSum, s);
        __syncthreads();
        if (threadIdx.x == 0) result = sSum + (float)(K - sCount) * tstar;
    }
    if (threadIdx.x == 0) atomicAdd(&scalars[2], result);
}

// ---------------- K6: finalize ----------------
__global__ void finalize_kernel(const unsigned* __restrict__ num_pos,
                                const float* __restrict__ scalars,
                                float* __restrict__ out)
{
    if (threadIdx.x == 0 && blockIdx.x == 0) {
        unsigned N = 0;
        for (int b = 0; b < B_; ++b) N += num_pos[b];
        float Nf = (float)N;
        out[0] = scalars[0] / Nf;                 // loss_l / N
        out[1] = (scalars[1] + scalars[2]) / Nf;  // loss_c / N
    }
}

extern "C" void kernel_launch(void* const* d_in, const int* in_sizes, int n_in,
                              void* d_out, int out_size, void* d_ws, size_t ws_size,
                              hipStream_t stream) {
    const float* arm_loc   = (const float*)d_in[0];
    const float* arm_conf  = (const float*)d_in[1];
    const float* odm_loc   = (const float*)d_in[2];
    const float* odm_conf  = (const float*)d_in[3];
    const float* priors    = (const float*)d_in[4];
    const float* gt_boxes  = (const float*)d_in[5];
    const int*   gt_labels = (const int*)d_in[6];
    float* out = (float*)d_out;

    const size_t BP = (size_t)B_ * P_;
    float* bto = (float*)d_ws;
    int*   bti = (int*)(bto + BP);
    int*   cft = (int*)(bti + BP);
    int*   psA = (int*)(cft + BP);
    float* cem = (float*)(psA + BP);
    unsigned long long* gBest = (unsigned long long*)(cem + BP);  // 8B aligned
    unsigned* npos = (unsigned*)(gBest + B_ * O_);
    float* scal = (float*)(npos + B_);  // [0]=loss_l [1]=pos_ce [2]=neg_sum

    const int zeroWords = B_ * O_ * 2 + B_ + 3;
    init_kernel<<<4, 256, 0, stream>>>((unsigned*)gBest, zeroWords);

    match_kernel<<<B_, 256, 0, stream>>>(arm_loc, priors, gt_boxes, bto, bti, gBest);
    override_kernel<<<1, 64, 0, stream>>>(gBest, bto, bti);
    pos_loc_kernel<<<4080, 256, 0, stream>>>(arm_loc, arm_conf, odm_loc, priors,
                                             gt_boxes, gt_labels, bto, bti,
                                             cft, psA, npos, scal);
    ce_wmma_kernel<<<2040, 256, 0, stream>>>(odm_conf, cft, psA, cem, scal);
    topk_kernel<<<B_, 256, 0, stream>>>(cem, npos, scal);
    finalize_kernel<<<1, 32, 0, stream>>>(npos, scal, out);
}
